// AttentionHyperNet_17231408792034
// MI455X (gfx1250) — matrix operand order repaired and checked by reading, not compiled, not run on hardware
//
#include <hip/hip_runtime.h>

// ---------- CDNA5 WMMA types ----------
typedef __attribute__((ext_vector_type(16))) __bf16 v16bf;
typedef __attribute__((ext_vector_type(8)))  float  v8f;

union BF16x16 { v16bf v; unsigned short s[16]; };
union BF16x16U4 { uint4 u[2]; v16bf v; };

// Native hardware f32 -> bf16 convert (v_cvt_pk_bf16_f32 on gfx1250)
__device__ __forceinline__ unsigned short f2bf(float f) {
  union { __bf16 b; unsigned short s; } u;
  u.b = (__bf16)f;
  return u.s;
}

__device__ __forceinline__ v8f zero8() {
  v8f z = {0.f, 0.f, 0.f, 0.f, 0.f, 0.f, 0.f, 0.f};
  return z;
}

__device__ __forceinline__ v8f wmma_bf16(v16bf a, v16bf b, v8f c) {
  return __builtin_amdgcn_wmma_f32_16x16x32_bf16(false, a, false, b, (short)0, c,
                                                 false, false);
}

// A-fragment (16x32 bf16, MxK) from row-major LDS, vectorized:
// lane's 16 elems = two contiguous 8-elem runs (16B each, 16B-aligned).
__device__ __forceinline__ v16bf load_a_lds16(const unsigned short* base, int stride,
                                              int m0, int k0, int half, int lr) {
  const unsigned short* p = base + (m0 + lr) * stride + k0 + half * 16;
  BF16x16U4 t;
  t.u[0] = *(const uint4*)(p);       // K = k0+16h .. +7
  t.u[1] = *(const uint4*)(p + 8);   // K = k0+16h+8 .. +15
  return t.v;
}

// B-fragment from fragment-native LDS layout: 32B contiguous per lane.
__device__ __forceinline__ v16bf load_frag_lds(const unsigned short* base,
                                               int frag, int lane) {
  const uint4* p = (const uint4*)(base + (frag * 32 + lane) * 16);
  BF16x16U4 t;
  t.u[0] = p[0];
  t.u[1] = p[1];
  return t.v;
}

// B-fragment from pre-swizzled bf16 weights in d_ws (two b128 global loads).
__device__ __forceinline__ v16bf load_b_frag(const unsigned short* __restrict__ ws,
                                             int Kt, int ks, int nt, int lane) {
  const uint4* p = (const uint4*)(ws + (((nt * Kt + ks) * 32) + lane) * 16);
  BF16x16U4 t;
  t.u[0] = p[0];
  t.u[1] = p[1];
  return t.v;
}

// ---------- workspace (pre-converted weights, ushort offsets) ----------
static const int WSU_W1   = 0;                     // 8*1*512  us
static const int WSU_QKV  = WSU_W1 + 4096;         // 24*4*512 us
static const int WSU_WOUT = WSU_QKV + 49152;       // 8*4*512  us
static const int WSU_W2   = WSU_WOUT + 16384;      // 2*4*512  us
static const int WSU_TOTAL = WSU_W2 + 4096;        // 73728 us = 147456 B

// ---------- weight pre-convert / pre-swizzle kernel ----------
__device__ __forceinline__ unsigned short conv_elem(const float* __restrict__ W,
                                                    int e, int Kt, int ldb, int kmax) {
  int tile = e >> 9;  // 512 elems per fragment (32 lanes x 16)
  int within = e & 511;
  int lane = within >> 4;
  int i = within & 15;
  int ks = tile % Kt;
  int nt = tile / Kt;
  int half = lane >> 4;
  int lr = lane & 15;
  int k = ks * 32 + i + 16 * half;
  int n = nt * 16 + lr;
  return (k < kmax) ? f2bf(W[k * ldb + n]) : (unsigned short)0;
}

extern "C" __global__ __launch_bounds__(256)
void weight_convert(const float* __restrict__ W1, const float* __restrict__ Wqkv,
                    const float* __restrict__ Wout, const float* __restrict__ W2,
                    unsigned short* __restrict__ ws) {
  int e = blockIdx.x * 256 + threadIdx.x;
  if (e < 4096) {
    ws[WSU_W1 + e] = conv_elem(W1, e, 1, 128, 19);
  } else if (e < 53248) {
    ws[WSU_QKV + (e - 4096)] = conv_elem(Wqkv, e - 4096, 4, 384, 128);
  } else if (e < 69632) {
    ws[WSU_WOUT + (e - 53248)] = conv_elem(Wout, e - 53248, 4, 128, 128);
  } else if (e < 73728) {
    ws[WSU_W2 + (e - 69632)] = conv_elem(W2, e - 69632, 4, 32, 128);
  }
}

// ---------- LDS layout (dynamic shared, bytes) ----------
#define X1S 136  // padded row stride (elems); 272B = 17*16B -> conflict-free b128
// region0: x1[128][136]   (A/B)  OVERLAPS  wbuf[4][64][136]   (C/D)
// region1: q[64][136]     (B/C)  OVERLAPS  attn[64][136]      (D/E)
// region2: kTfrag 32*512us(B/C)  OVERLAPS  x2[64][136]        (E/F)
// region3: vfrag  32*512us(B/D)  OVERLAPS  ent f32[128*19]    (pre-A/A)
// region4: em[128] floats
static const int OFF_W  = 0;
static const int OFF_Q  = OFF_W  + 4 * 64 * X1S * 2;   // 69632
static const int OFF_KT = OFF_Q  + 64 * X1S * 2;       // 87040
static const int OFF_V  = OFF_KT + 32 * 512 * 2;       // 119808
static const int OFF_EM = OFF_V  + 32 * 512 * 2;       // 152576
static const int SMEM_BYTES = OFF_EM + 128 * 4;        // 153088 B

extern "C" __global__ __launch_bounds__(256)
void attn_hypernet_fused(const float* __restrict__ entities,
                         const int*   __restrict__ entity_mask,
                         const float* __restrict__ b1,
                         const float* __restrict__ bout,
                         const float* __restrict__ b2,
                         const unsigned short* __restrict__ ws,
                         float* __restrict__ out) {
  extern __shared__ unsigned char smem[];
  unsigned short* x1buf  = (unsigned short*)(smem + OFF_W);
  unsigned short* wbuf   = (unsigned short*)(smem + OFF_W);
  unsigned short* qbuf   = (unsigned short*)(smem + OFF_Q);
  unsigned short* attnb  = (unsigned short*)(smem + OFF_Q);
  unsigned short* ktfrag = (unsigned short*)(smem + OFF_KT);
  unsigned short* x2buf  = (unsigned short*)(smem + OFF_KT);
  unsigned short* vfrag  = (unsigned short*)(smem + OFF_V);
  float*          entf   = (float*)(smem + OFF_V);  // dead by phase B
  float* emf = (float*)(smem + OFF_EM);

  const unsigned short* wsW1   = ws + WSU_W1;
  const unsigned short* wsQKV  = ws + WSU_QKV;
  const unsigned short* wsWOUT = ws + WSU_WOUT;
  const unsigned short* wsW2   = ws + WSU_W2;

  const int bidx = blockIdx.x;
  const int tid  = threadIdx.x;
  const int wv   = tid >> 5;
  const int lane = tid & 31;
  const int half = lane >> 4;
  const int lr   = lane & 15;
  const float NEG_INF = -__builtin_inff();

  // stage masks + entities tile (coalesced)
  if (tid < 128) emf[tid] = (float)entity_mask[bidx * 128 + tid];
  {
    const float* eb = entities + (long)bidx * 128 * 19;
    for (int i = tid; i < 128 * 19; i += 256) entf[i] = eb[i];
  }
  __syncthreads();

  // ===== Phase A: x1 = relu(entities @ W1 + b1); M=128,N=128,K=19->32 =====
  {
    const int mt = wv;
    BF16x16 ta;
    const int row = mt * 16 + lr;
#pragma unroll
    for (int i = 0; i < 16; ++i) {
      int k = i + half * 16;
      ta.s[i] = (k < 19) ? f2bf(entf[row * 19 + k]) : (unsigned short)0;
    }
    for (int nt = 0; nt < 8; ++nt) {
      v16bf tb = load_b_frag(wsW1, 1, 0, nt, lane);
      v8f c = zero8();
      c = wmma_bf16(ta.v, tb, c);
      const int col = nt * 16 + lr;
      const float bn = b1[col];
#pragma unroll
      for (int r = 0; r < 8; ++r) {
        int m = mt * 16 + r + half * 8;
        float v0 = c[r] + bn;
        x1buf[m * X1S + col] = f2bf(v0 > 0.f ? v0 : 0.f);
      }
    }
  }
  __syncthreads();

  // ===== Phase B: qkv = x1 @ Wqkv; scatter to q (row-major), kT/v (frag layout) =====
  {
    const int mt = wv;
    v16bf af[4];
#pragma unroll
    for (int ks = 0; ks < 4; ++ks)
      af[ks] = load_a_lds16(x1buf, X1S, mt * 16, ks * 32, half, lr);
    for (int nt = 0; nt < 24; ++nt) {
      if (nt < 8 && mt >= 4) continue;  // q rows >= 64 are never used
      v8f c = zero8();
#pragma unroll
      for (int ks = 0; ks < 4; ++ks) {
        v16bf tb = load_b_frag(wsQKV, 4, ks, nt, lane);
        c = wmma_bf16(af[ks], tb, c);
      }
      const int gc = nt * 16;
#pragma unroll
      for (int r = 0; r < 8; ++r) {
        int m = mt * 16 + r + half * 8;
        int n = gc + lr;
        unsigned short bv = f2bf(c[r]);
        if (gc < 128) {
          // q row-major: [agent][E]
          qbuf[m * X1S + n] = bv;
        } else if (gc < 256) {
          // kT fragment layout: frag(h,ntK=m>>4), k-dim = d, col = key
          int cc = n - 128; int h = cc >> 5, d = cc & 31;
          int ntK = m >> 4, lrK = m & 15;
          ktfrag[(((h * 8 + ntK) * 32) + (d >> 4) * 16 + lrK) * 16 + (d & 15)] = bv;
        } else {
          // v fragment layout: frag(h, ks=m>>5, ntV=d>>4), k-dim = key, col = dim
          int cc = n - 256; int h = cc >> 5, d = cc & 31;
          int ksV = m >> 5, kr = m & 31;
          vfrag[(((h * 8 + ksV * 2 + (d >> 4)) * 32) + (kr >> 4) * 16 + (d & 15)) * 16 +
                (kr & 15)] = bv;
        }
      }
    }
  }
  __syncthreads();

  // ===== Phase C: logits = q.kT/sqrt(hd), mask, in-register softmax =====
  for (int unit = wv; unit < 16; unit += 8) {
    const int h = unit >> 2, mt = unit & 3;
    v16bf aq = load_a_lds16(qbuf, X1S, mt * 16, h * 32, half, lr);
    float cfr[8][8];
    float emm[8];
#pragma unroll
    for (int r = 0; r < 8; ++r) emm[r] = emf[mt * 16 + r + half * 8];
    const float scale = 0.17677669529663687f;  // 1/sqrt(32)
    for (int nt = 0; nt < 8; ++nt) {
      v16bf bk = load_frag_lds(ktfrag, h * 8 + nt, lane);
      v8f c = zero8();
      c = wmma_bf16(aq, bk, c);
      float emj = emf[nt * 16 + lr];
#pragma unroll
      for (int r = 0; r < 8; ++r)
        cfr[nt][r] = (emm[r] > 0.f || emj > 0.f) ? NEG_INF : c[r] * scale;
    }
#pragma unroll
    for (int r = 0; r < 8; ++r) {
      float mx = cfr[0][r];
#pragma unroll
      for (int t = 1; t < 8; ++t) mx = fmaxf(mx, cfr[t][r]);
#pragma unroll
      for (int s = 1; s <= 8; s <<= 1) mx = fmaxf(mx, __shfl_xor(mx, s, 32));
      float sum = 0.f;
#pragma unroll
      for (int t = 0; t < 8; ++t) {
        float e = (mx > NEG_INF) ? __expf(cfr[t][r] - mx) : 0.f;
        cfr[t][r] = e;
        sum += e;
      }
#pragma unroll
      for (int s = 1; s <= 8; s <<= 1) sum += __shfl_xor(sum, s, 32);
      float inv = (sum > 0.f) ? 1.f / sum : 0.f;  // NaN -> 0 semantics
      int m = mt * 16 + r + half * 8;
#pragma unroll
      for (int t = 0; t < 8; ++t)
        wbuf[(h * 64 + m) * X1S + t * 16 + lr] = f2bf(cfr[t][r] * inv);
    }
  }
  __syncthreads();

  // ===== Phase D: attn = w @ v per head; M=64,N=32,K=128 =====
  for (int unit = wv; unit < 16; unit += 8) {
    const int h = unit >> 2, mt = unit & 3;
    v16bf aw[4];
#pragma unroll
    for (int ks = 0; ks < 4; ++ks)
      aw[ks] = load_a_lds16(wbuf + h * 64 * X1S, X1S, mt * 16, ks * 32, half, lr);
#pragma unroll
    for (int nt = 0; nt < 2; ++nt) {
      v8f c = zero8();
#pragma unroll
      for (int ks = 0; ks < 4; ++ks) {
        v16bf bv16 = load_frag_lds(vfrag, h * 8 + ks * 2 + nt, lane);
        c = wmma_bf16(aw[ks], bv16, c);
      }
#pragma unroll
      for (int r = 0; r < 8; ++r) {
        int m = mt * 16 + r + half * 8;
        attnb[m * X1S + h * 32 + nt * 16 + lr] = f2bf(c[r]);
      }
    }
  }
  __syncthreads();

  // ===== Phase E: x2 = attn @ Wout + bout, masked; M=64,N=128,K=128 =====
  {
    const int mt = wv & 3;
    const int nt0 = (wv >> 2) * 4;
    v16bf aa[4];
#pragma unroll
    for (int ks = 0; ks < 4; ++ks)
      aa[ks] = load_a_lds16(attnb, X1S, mt * 16, ks * 32, half, lr);
    for (int nt = nt0; nt < nt0 + 4; ++nt) {
      v8f c = zero8();
#pragma unroll
      for (int ks = 0; ks < 4; ++ks) {
        v16bf tb = load_b_frag(wsWOUT, 4, ks, nt, lane);
        c = wmma_bf16(aa[ks], tb, c);
      }
      const int n = nt * 16 + lr;
      const float bo = bout[n];
#pragma unroll
      for (int r = 0; r < 8; ++r) {
        int m = mt * 16 + r + half * 8;
        float v0 = c[r] + bo;
        if (emf[m] > 0.f) v0 = 0.f;
        x2buf[m * X1S + n] = f2bf(v0);
      }
    }
  }
  __syncthreads();

  // ===== Phase F: x3 = x2 @ W2 + b2, masked; M=64,N=32,K=128 =====
  {
    const int mt = wv >> 1, nt = wv & 1;
    v8f c = zero8();
#pragma unroll
    for (int ks = 0; ks < 4; ++ks) {
      v16bf a = load_a_lds16(x2buf, X1S, mt * 16, ks * 32, half, lr);
      v16bf tb = load_b_frag(wsW2, 4, ks, nt, lane);
      c = wmma_bf16(a, tb, c);
    }
    const int n = nt * 16 + lr;
    const float bb = b2[n];
#pragma unroll
    for (int r = 0; r < 8; ++r) {
      int m = mt * 16 + r + half * 8;
      float v0 = c[r] + bb;
      if (emf[m] > 0.f) v0 = 0.f;
      out[((long)bidx * 64 + m) * 32 + n] = v0;
    }
  }
}

extern "C" void kernel_launch(void* const* d_in, const int* in_sizes, int n_in,
                              void* d_out, int out_size, void* d_ws, size_t ws_size,
                              hipStream_t stream) {
  const float* entities    = (const float*)d_in[0];
  const int*   entity_mask = (const int*)d_in[1];
  const float* W1   = (const float*)d_in[2];
  const float* b1   = (const float*)d_in[3];
  const float* Wqkv = (const float*)d_in[4];
  const float* Wout = (const float*)d_in[5];
  const float* bout = (const float*)d_in[6];
  const float* W2   = (const float*)d_in[7];
  const float* b2   = (const float*)d_in[8];
  float* out = (float*)d_out;
  unsigned short* ws = (unsigned short*)d_ws;

  hipLaunchKernelGGL(weight_convert, dim3((WSU_TOTAL + 255) / 256), dim3(256), 0,
                     stream, W1, Wqkv, Wout, W2, ws);

  const int bs = in_sizes[0] / (128 * 19);
  hipLaunchKernelGGL(attn_hypernet_fused, dim3(bs), dim3(256), SMEM_BYTES, stream,
                     entities, entity_mask, b1, bout, b2, ws, out);
}
